// Net_18210661335121
// MI455X (gfx1250) — matrix operand order repaired and checked by reading, not compiled, not run on hardware
//
#include <hip/hip_runtime.h>
#include <hip/hip_bf16.h>

// ---------------------------------------------------------------------------
// Problem constants (match reference)
// ---------------------------------------------------------------------------
#define IN_FEATS 64
#define N_NODES  50000
#define N_EDGES  800000
#define N_GRAPHS 64
#define EDGE_DIM 10
#define MSG_IN   (2 * IN_FEATS + EDGE_DIM)   // 138
#define H1       (4 * IN_FEATS)              // 256
#define H2       (2 * IN_FEATS)              // 128
#define PWIDTH   (4 * IN_FEATS)              // 256 projected feats per node
#define EA       (N_EDGES + N_NODES)         // edges + self loops
// P row padding: 391 blocks * 8 waves * 16 rows = 50048 (stores unconditional)
#define NP_ROWS  (((N_NODES + 127) / 128) * 128)
// padded LDS stride for 10-wide dist weights (stride 11 -> conflict-free banks)
#define WPAD     11

typedef __attribute__((ext_vector_type(16))) __bf16 v16bf;
typedef __attribute__((ext_vector_type(8)))  float  v8f;

union Frag16 { v16bf v; uint4 q[2]; };

// round-to-nearest-even f32 -> bf16 (bit pattern in ushort)
__device__ __forceinline__ unsigned short f2bf(float f) {
    unsigned int u = __float_as_uint(f);
    unsigned int r = u + 0x7FFFu + ((u >> 16) & 1u);
    return (unsigned short)(r >> 16);
}

__device__ __forceinline__ float softplusf(float x) {
    return x > 20.f ? x : log1pf(__expf(x));
}
__device__ __forceinline__ float sigmoidf_(float x) {
    return 1.f / (1.f + __expf(-x));
}

// ---------------------------------------------------------------------------
// zero-fill helper
// ---------------------------------------------------------------------------
__global__ void zero_kernel(float* p, int n) {
    int t = blockIdx.x * blockDim.x + threadIdx.x;
    if (t < n) p[t] = 0.f;
}

// ---------------------------------------------------------------------------
// f32 -> bf16 conversion of env
// ---------------------------------------------------------------------------
__global__ void convert_bf16_kernel(const float* __restrict__ env,
                                    unsigned short* __restrict__ envb, int n) {
    int t = blockIdx.x * blockDim.x + threadIdx.x;
    if (t < n) envb[t] = f2bf(env[t]);
}

// ---------------------------------------------------------------------------
// Pack this layer's weights into fragment-ready bf16 B-tiles.
// Wcat[j][k] (j in 0..255, k in 0..63):
//   j<64: Wf[j][k]; j<128: Wf[j-64][64+k]; j<192: Ws[j-128][k]; else Ws[j-192][64+k]
// Fragment layout for WMMA B (32x16 bf16) tile (jt, kc):
//   lane<16 : col = jt*16+lane,    k = kc*32 + h       (h = 0..15)
//   lane>=16: col = jt*16+lane-16, k = kc*32 + 16 + h
// Bpack index: ((jt*2+kc)*32 + lane)*16 + h  -> 16384 halves total
// ---------------------------------------------------------------------------
__global__ void pack_weights_kernel(const float* __restrict__ Wf,
                                    const float* __restrict__ Ws,
                                    unsigned short* __restrict__ Bpack) {
    int idx = blockIdx.x * blockDim.x + threadIdx.x;   // 0..16383
    if (idx >= 16 * 2 * 32 * 16) return;
    int h    = idx & 15;
    int lane = (idx >> 4) & 31;
    int kc   = (idx >> 9) & 1;
    int jt   = idx >> 10;
    int col  = jt * 16 + (lane & 15);
    int k    = kc * 32 + ((lane >> 4) ? 16 : 0) + h;
    float v;
    if (col < 64)        v = Wf[(size_t)col * MSG_IN + k];
    else if (col < 128)  v = Wf[(size_t)(col - 64) * MSG_IN + 64 + k];
    else if (col < 192)  v = Ws[(size_t)(col - 128) * MSG_IN + k];
    else                 v = Ws[(size_t)(col - 192) * MSG_IN + 64 + k];
    Bpack[idx] = f2bf(v);
}

// ---------------------------------------------------------------------------
// Node projection: P[NP_ROWS,256] = env_bf16[N,64] @ Wcat^T
// via v_wmma_f32_16x16x32_bf16 (K=64 -> 2 chained WMMAs per tile).
// Block = 256 threads = 8 waves; wave owns one 16-row tile, sweeps 16 col
// tiles. B fragments staged in LDS. Stores are unconditional (P is padded),
// so the inner loop is branch-free and EXEC stays all-ones throughout.
// ---------------------------------------------------------------------------
__global__ void node_proj_kernel(const unsigned short* __restrict__ envb,
                                 const unsigned short* __restrict__ Bpack,
                                 float* __restrict__ P, int nrows) {
    __shared__ unsigned short sB[16384];            // 32 KB fragment-ready weights
    {
        uint4* dstq = (uint4*)sB;
        const uint4* srcq = (const uint4*)Bpack;
        for (int i = threadIdx.x; i < 2048; i += blockDim.x) dstq[i] = srcq[i];
    }
    __syncthreads();

    const int wave = threadIdx.x >> 5;
    const int lane = threadIdx.x & 31;
    const int m    = lane & 15;
    const int sel  = lane >> 4;

    const int nbase = (blockIdx.x * 8 + wave) * 16;

    const int row = min(nbase + m, nrows - 1);       // clamp keeps EXEC full
    const unsigned short* er = envb + (size_t)row * IN_FEATS;

    // A fragments (contiguous K, matching Bpack): lane<16 holds {0..15},
    // lane>=16 holds {16..31}, per 32-wide K chunk.
    Frag16 a[2];
#pragma unroll
    for (int kc = 0; kc < 2; ++kc) {
        int off0 = kc * 32 + sel * 16;
        int off1 = kc * 32 + sel * 16 + 8;
        a[kc].q[0] = *(const uint4*)(er + off0);
        a[kc].q[1] = *(const uint4*)(er + off1);
    }

    // output row block for this lane-half; P is padded to NP_ROWS rows
    float* prow = P + (size_t)(nbase + (sel ? 8 : 0)) * PWIDTH + m;

#pragma unroll
    for (int jt = 0; jt < 16; ++jt) {
        v8f c = {};
#pragma unroll
        for (int kc = 0; kc < 2; ++kc) {
            Frag16 b;
            const uint4* bp = (const uint4*)(sB + ((size_t)(jt * 2 + kc) * 32 + lane) * 16);
            b.q[0] = bp[0];
            b.q[1] = bp[1];
            c = __builtin_amdgcn_wmma_f32_16x16x32_bf16(
                    false, a[kc].v, false, b.v, (short)0, c, false, false);
        }
        // C/D layout: VGPR r -> row M = r + (sel?8:0), col N = lane&15
        float* pc = prow + jt * 16;
#pragma unroll
        for (int r = 0; r < 8; ++r) {
            pc[(size_t)r * PWIDTH] = c[r];
        }
    }
}

// ---------------------------------------------------------------------------
// Edge pass: 8 lanes per edge, 8 feats per lane.
//   af = P[s][f] + P[d][64+f] + bf[f] + dist . Wf_c[f]
//   as = P[s][128+f] + P[d][192+f] + bs[f] + dist . Ws_c[f]
//   env[d][f] += sigmoid(af) * softplus(as)      (atomic f32)
// Self loops (e >= N_EDGES): src=dst=e-N_EDGES, dist = 0.
// ---------------------------------------------------------------------------
__global__ void edge_msg_kernel(const float* __restrict__ P,
                                const float* __restrict__ dist,
                                const int* __restrict__ src,
                                const int* __restrict__ dst,
                                const float* __restrict__ Wf,
                                const float* __restrict__ Ws,
                                const float* __restrict__ bf,
                                const float* __restrict__ bs,
                                float* __restrict__ env) {
    __shared__ float wcf[IN_FEATS][WPAD];
    __shared__ float wcs[IN_FEATS][WPAD];
    __shared__ float bfv[IN_FEATS];
    __shared__ float bsv[IN_FEATS];

    for (int i = threadIdx.x; i < IN_FEATS * EDGE_DIM; i += blockDim.x) {
        int r = i / EDGE_DIM, cidx = i % EDGE_DIM;
        wcf[r][cidx] = Wf[(size_t)r * MSG_IN + 2 * IN_FEATS + cidx];
        wcs[r][cidx] = Ws[(size_t)r * MSG_IN + 2 * IN_FEATS + cidx];
    }
    if (threadIdx.x < IN_FEATS) {
        bfv[threadIdx.x] = bf[threadIdx.x];
        bsv[threadIdx.x] = bs[threadIdx.x];
    }
    __syncthreads();

    int gt  = blockIdx.x * blockDim.x + threadIdx.x;
    int e   = gt >> 3;
    int f0  = (gt & 7) * 8;
    if (e >= EA) return;

    int s, d;
    const float* dp = nullptr;
    if (e < N_EDGES) { s = src[e]; d = dst[e]; dp = dist + (size_t)e * EDGE_DIM; }
    else             { s = d = e - N_EDGES; }

    const float* Ps = P + (size_t)s * PWIDTH;
    const float* Pd = P + (size_t)d * PWIDTH;
    float4 a0 = *(const float4*)(Ps + f0);
    float4 a1 = *(const float4*)(Ps + f0 + 4);
    float4 b0 = *(const float4*)(Pd + 64 + f0);
    float4 b1 = *(const float4*)(Pd + 64 + f0 + 4);
    float4 c0 = *(const float4*)(Ps + 128 + f0);
    float4 c1 = *(const float4*)(Ps + 128 + f0 + 4);
    float4 d0 = *(const float4*)(Pd + 192 + f0);
    float4 d1 = *(const float4*)(Pd + 192 + f0 + 4);

    float af[8], as_[8];
    af[0] = a0.x + b0.x; af[1] = a0.y + b0.y; af[2] = a0.z + b0.z; af[3] = a0.w + b0.w;
    af[4] = a1.x + b1.x; af[5] = a1.y + b1.y; af[6] = a1.z + b1.z; af[7] = a1.w + b1.w;
    as_[0] = c0.x + d0.x; as_[1] = c0.y + d0.y; as_[2] = c0.z + d0.z; as_[3] = c0.w + d0.w;
    as_[4] = c1.x + d1.x; as_[5] = c1.y + d1.y; as_[6] = c1.z + d1.z; as_[7] = c1.w + d1.w;
#pragma unroll
    for (int i = 0; i < 8; ++i) { af[i] += bfv[f0 + i]; as_[i] += bsv[f0 + i]; }

    if (dp) {
        float dv[EDGE_DIM];
#pragma unroll
        for (int k = 0; k < EDGE_DIM; ++k) dv[k] = dp[k];
#pragma unroll
        for (int i = 0; i < 8; ++i) {
            float accf = 0.f, accs = 0.f;
#pragma unroll
            for (int k = 0; k < EDGE_DIM; ++k) {
                accf = fmaf(dv[k], wcf[f0 + i][k], accf);
                accs = fmaf(dv[k], wcs[f0 + i][k], accs);
            }
            af[i] += accf; as_[i] += accs;
        }
    }

    float* ed = env + (size_t)d * IN_FEATS + f0;
#pragma unroll
    for (int i = 0; i < 8; ++i) {
        float mval = sigmoidf_(af[i]) * softplusf(as_[i]);
        atomicAdd(ed + i, mval);
    }
}

// ---------------------------------------------------------------------------
// Pooling: per-graph sums + counts (atomic; gid values 0..63)
// ---------------------------------------------------------------------------
__global__ void pool_feats_kernel(const float* __restrict__ env,
                                  const int* __restrict__ gid,
                                  float* __restrict__ pooled) {
    int t = blockIdx.x * blockDim.x + threadIdx.x;
    if (t >= N_NODES * IN_FEATS) return;
    int n = t >> 6, f = t & 63;
    atomicAdd(pooled + (size_t)gid[n] * IN_FEATS + f, env[t]);
}

__global__ void pool_counts_kernel(const int* __restrict__ gid,
                                   float* __restrict__ counts) {
    int n = blockIdx.x * blockDim.x + threadIdx.x;
    if (n < N_NODES) atomicAdd(counts + gid[n], 1.f);
}

// ---------------------------------------------------------------------------
// Readout MLP (tiny: 64 graphs)
// ---------------------------------------------------------------------------
__global__ void mlp1_kernel(const float* __restrict__ pooled,
                            const float* __restrict__ counts,
                            const float* __restrict__ W21,
                            const float* __restrict__ b21,
                            float* __restrict__ h1) {
    __shared__ float x[IN_FEATS];
    int g = blockIdx.x;
    if (threadIdx.x < IN_FEATS) {
        float cnt = fmaxf(counts[g], 1.f);
        x[threadIdx.x] = pooled[(size_t)g * IN_FEATS + threadIdx.x] / cnt;
    }
    __syncthreads();
    int j = threadIdx.x;                      // 0..255
    float acc = b21[j];
    const float* wr = W21 + (size_t)j * IN_FEATS;
#pragma unroll 8
    for (int k = 0; k < IN_FEATS; ++k) acc = fmaf(x[k], wr[k], acc);
    h1[(size_t)g * H1 + j] = softplusf(acc);
}

__global__ void mlp2_kernel(const float* __restrict__ h1,
                            const float* __restrict__ W22,
                            const float* __restrict__ b22,
                            float* __restrict__ h2) {
    __shared__ float x[H1];
    int g = blockIdx.x;
    x[threadIdx.x]       = h1[(size_t)g * H1 + threadIdx.x];
    x[threadIdx.x + 128] = h1[(size_t)g * H1 + threadIdx.x + 128];
    __syncthreads();
    int j = threadIdx.x;                      // 0..127
    float acc = b22[j];
    const float* wr = W22 + (size_t)j * H1;
#pragma unroll 8
    for (int k = 0; k < H1; ++k) acc = fmaf(x[k], wr[k], acc);
    h2[(size_t)g * H2 + j] = softplusf(acc);
}

__global__ void mlp3_kernel(const float* __restrict__ h2,
                            const float* __restrict__ W23,
                            const float* __restrict__ b23,
                            float* __restrict__ out) {
    int g = threadIdx.x;                      // 0..63
    if (g >= N_GRAPHS) return;
    float acc = b23[0];
    const float* xr = h2 + (size_t)g * H2;
#pragma unroll 8
    for (int k = 0; k < H2; ++k) acc = fmaf(xr[k], W23[k], acc);
    out[g] = acc;
}

// ---------------------------------------------------------------------------
// Host launcher
// ---------------------------------------------------------------------------
static inline size_t alignup(size_t x) { return (x + 255) & ~size_t(255); }

extern "C" void kernel_launch(void* const* d_in, const int* in_sizes, int n_in,
                              void* d_out, int out_size, void* d_ws, size_t ws_size,
                              hipStream_t stream) {
    (void)in_sizes; (void)n_in; (void)out_size; (void)ws_size;

    const float* Z    = (const float*)d_in[0];
    const float* dist = (const float*)d_in[1];
    const int*   src  = (const int*)d_in[2];
    const int*   dst  = (const int*)d_in[3];
    const int*   gid  = (const int*)d_in[4];
    // layer weights: d_in[5 + 4*L + {0:Wf,1:bf,2:Ws,3:bs}]
    const float* W21 = (const float*)d_in[17];
    const float* b21 = (const float*)d_in[18];
    const float* W22 = (const float*)d_in[19];
    const float* b22 = (const float*)d_in[20];
    const float* W23 = (const float*)d_in[21];
    const float* b23 = (const float*)d_in[22];
    float* out = (float*)d_out;

    // workspace carve-out
    char* w = (char*)d_ws;
    float*          env    = (float*)w;          w += alignup((size_t)N_NODES * IN_FEATS * 4);
    unsigned short* envb   = (unsigned short*)w; w += alignup((size_t)N_NODES * IN_FEATS * 2);
    float*          P      = (float*)w;          w += alignup((size_t)NP_ROWS * PWIDTH * 4);
    unsigned short* Bpack  = (unsigned short*)w; w += alignup((size_t)16384 * 2);
    float*          pooled = (float*)w;          w += alignup((size_t)N_GRAPHS * IN_FEATS * 4);
    float*          counts = (float*)w;          w += alignup((size_t)N_GRAPHS * 4);
    float*          h1     = (float*)w;          w += alignup((size_t)N_GRAPHS * H1 * 4);
    float*          h2     = (float*)w;          w += alignup((size_t)N_GRAPHS * H2 * 4);

    // env <- Z
    hipMemcpyAsync(env, Z, (size_t)N_NODES * IN_FEATS * sizeof(float),
                   hipMemcpyDeviceToDevice, stream);

    // zero pooled + counts
    zero_kernel<<<(N_GRAPHS * IN_FEATS + 255) / 256, 256, 0, stream>>>(pooled, N_GRAPHS * IN_FEATS);
    zero_kernel<<<1, 256, 0, stream>>>(counts, N_GRAPHS);

    const int nelem   = N_NODES * IN_FEATS;
    const int cvtBlk  = (nelem + 255) / 256;
    const int projBlk = NP_ROWS / 128;                           // 8 row-tiles / block
    const int edgeBlk = ((EA * 8) + 255) / 256;

    for (int L = 0; L < 3; ++L) {
        const float* Wf = (const float*)d_in[5 + 4 * L + 0];
        const float* bf = (const float*)d_in[5 + 4 * L + 1];
        const float* Ws = (const float*)d_in[5 + 4 * L + 2];
        const float* bs = (const float*)d_in[5 + 4 * L + 3];

        convert_bf16_kernel<<<cvtBlk, 256, 0, stream>>>(env, envb, nelem);
        pack_weights_kernel<<<64, 256, 0, stream>>>(Wf, Ws, Bpack);
        node_proj_kernel<<<projBlk, 256, 0, stream>>>(envb, Bpack, P, N_NODES);
        edge_msg_kernel<<<edgeBlk, 256, 0, stream>>>(P, dist, src, dst, Wf, Ws, bf, bs, env);
    }

    pool_feats_kernel<<<(nelem + 255) / 256, 256, 0, stream>>>(env, gid, pooled);
    pool_counts_kernel<<<(N_NODES + 255) / 256, 256, 0, stream>>>(gid, counts);

    mlp1_kernel<<<N_GRAPHS, H1, 0, stream>>>(pooled, counts, W21, b21, h1);
    mlp2_kernel<<<N_GRAPHS, H2, 0, stream>>>(h1, W22, b22, h2);
    mlp3_kernel<<<1, 64, 0, stream>>>(h2, W23, b23, out);
}
